// RNN_42949673216
// MI455X (gfx1250) — matrix operand order repaired and checked by reading, not compiled, not run on hardware
//
#include <hip/hip_runtime.h>
#include <hip/hip_bf16.h>

// Problem constants (match reference)
#define kT 512
#define kB 64
#define kI 512
#define kH 1024

typedef __bf16 bf16_t;
typedef __attribute__((ext_vector_type(16))) __bf16 v16bf;
typedef __attribute__((ext_vector_type(8)))  __bf16 v8bf;
typedef __attribute__((ext_vector_type(8)))  float  v8f;
typedef __attribute__((ext_vector_type(8)))  float  f32x8;

// ---------------------------------------------------------------------------
// Prep: convert+transpose f32 [K][N] -> bf16 [N][K] (so WMMA B-fragments are
// contiguous 32B per lane). One-time cost, bandwidth-trivial.
// ---------------------------------------------------------------------------
__global__ void cvt_transpose_kernel(const float* __restrict__ src,
                                     bf16_t* __restrict__ dst,
                                     int K, int N) {
    int idx = blockIdx.x * blockDim.x + threadIdx.x;
    if (idx < K * N) {
        int k = idx / N;
        int n = idx - k * N;
        dst[(size_t)n * K + k] = (bf16_t)src[idx];
    }
}

__global__ void cvt_kernel(const float* __restrict__ src,
                           bf16_t* __restrict__ dst, int n) {
    int idx = blockIdx.x * blockDim.x + threadIdx.x;
    if (idx < n) dst[idx] = (bf16_t)src[idx];
}

// ---------------------------------------------------------------------------
// Phase 1: xw = x @ Wx + b   (M = T*B = 32768, K = 512, N = 1024)
// One 16x16 tile per wave, 8 waves/block. A (x, f32, streamed/NT) converted
// to bf16 in registers; B from pre-transposed WxT [N][K] (reused, RT).
//
// 16-bit A 16x32 layout used: lane%16 -> M, per-lane K = khalf*16 + 0..15 (contig)
//   with khalf = lane>>4  (lanes 0-15: K 0-15 ; lanes 16-31: K 16-31)
// 16-bit B 32x16 layout used: lane%16 -> N, per-lane K = khalf*16 + 0..15 (contig)
// C/D f32: VGPR r, lane l -> M = r + (l>=16 ? 8 : 0), N = l&15
// ---------------------------------------------------------------------------
__global__ void __launch_bounds__(256)
xw_gemm_kernel(const float* __restrict__ x,     // [T*B, kI]
               const bf16_t* __restrict__ WxT,  // [kH, kI]
               const float* __restrict__ bias,  // [kH]
               float* __restrict__ xw) {        // [T*B, kH]
    const int lane  = threadIdx.x & 31;
    const int wave  = threadIdx.x >> 5;
    const int gw    = blockIdx.x * 8 + wave;
    const int tilesN = kH / 16;                 // 64
    const int tileM  = gw / tilesN;
    const int tileN  = gw - tileM * tilesN;
    const int khalf  = lane >> 4;
    const int lan    = lane & 15;

    const float*  arow = x   + (size_t)(tileM * 16 + lan) * kI;
    const bf16_t* bcol = WxT + (size_t)(tileN * 16 + lan) * kI;

    v8f acc = {};
    for (int k0 = 0; k0 < kI; k0 += 32) {
        // x is read exactly once across the whole run -> non-temporal loads
        f32x8 a0 = __builtin_nontemporal_load((const f32x8*)(arow + k0 + khalf * 16));
        f32x8 a1 = __builtin_nontemporal_load((const f32x8*)(arow + k0 + khalf * 16 + 8));
        v16bf a;
#pragma unroll
        for (int i = 0; i < 8; ++i) {
            a[i]     = (bf16_t)a0[i];
            a[i + 8] = (bf16_t)a1[i];
        }
        v16bf b = *(const v16bf*)(bcol + k0 + khalf * 16);
        __builtin_prefetch((const void*)(bcol + k0 + 32 + khalf * 16), 0, 1);
        acc = __builtin_amdgcn_wmma_f32_16x16x32_bf16(
            false, a, false, b, (short)0, acc, false, false);
    }

    const int n  = tileN * 16 + lan;
    const float bn = bias[n];
#pragma unroll
    for (int r = 0; r < 8; ++r) {
        int m = tileM * 16 + r + (khalf ? 8 : 0);
        xw[(size_t)m * kH + n] = acc[r] + bn;
    }
}

// ---------------------------------------------------------------------------
// Phase 2 (per timestep): h_new = tanh(xw_t + h @ Wh)
// M = 64, K = 1024, N = 1024 -> 256 tiles, 8 waves/block, 32 blocks.
// h kept as bf16 ping-pong in workspace; f32 result written straight into
// the outs[t] slice of d_out (NT store: write-once stream). WhT (2 MB bf16)
// stays L2-resident across steps; xw_t is read exactly once -> NT load.
// ---------------------------------------------------------------------------
__global__ void __launch_bounds__(256)
rnn_step_kernel(const float* __restrict__ xw_t,   // [kB, kH] slice
                const bf16_t* __restrict__ WhT,   // [kH, kH] (transposed bf16)
                const bf16_t* __restrict__ h_in,  // [kB, kH] bf16
                bf16_t* __restrict__ h_out,       // [kB, kH] bf16
                float* __restrict__ out_t) {      // [kB, kH] f32 (outs[t])
    const int lane  = threadIdx.x & 31;
    const int wave  = threadIdx.x >> 5;
    const int gw    = blockIdx.x * 8 + wave;
    const int tilesN = kH / 16;                  // 64
    const int tileM  = gw / tilesN;              // 0..3
    const int tileN  = gw - tileM * tilesN;
    const int khalf  = lane >> 4;
    const int lan    = lane & 15;

    const bf16_t* arow = h_in + (size_t)(tileM * 16 + lan) * kH;
    const bf16_t* bcol = WhT  + (size_t)(tileN * 16 + lan) * kH;

    v8f acc = {};
    for (int k0 = 0; k0 < kH; k0 += 32) {
        v8bf a0 = *(const v8bf*)(arow + k0 + khalf * 16);
        v8bf a1 = *(const v8bf*)(arow + k0 + khalf * 16 + 8);
        v16bf a;
#pragma unroll
        for (int i = 0; i < 8; ++i) {
            a[i]     = a0[i];
            a[i + 8] = a1[i];
        }
        v16bf b = *(const v16bf*)(bcol + k0 + khalf * 16);
        __builtin_prefetch((const void*)(bcol + k0 + 32 + khalf * 16), 0, 1);
        acc = __builtin_amdgcn_wmma_f32_16x16x32_bf16(
            false, a, false, b, (short)0, acc, false, false);
    }

    const int n = tileN * 16 + lan;
#pragma unroll
    for (int r = 0; r < 8; ++r) {
        int m = tileM * 16 + r + (khalf ? 8 : 0);
        size_t idx = (size_t)m * kH + n;
        float xv = __builtin_nontemporal_load(xw_t + idx);
        float v  = tanhf(acc[r] + xv);
        __builtin_nontemporal_store(v, out_t + idx);
        h_out[idx] = (bf16_t)v;
    }
}

// ---------------------------------------------------------------------------
// Launch. Inputs: x [T,B,I], h0 [B,H], Wx [I,H], Wh [H,H], b [1,H] (all f32).
// Output: outs [T,B,H] ++ h_last [B,H] (f32).
// ---------------------------------------------------------------------------
extern "C" void kernel_launch(void* const* d_in, const int* in_sizes, int n_in,
                              void* d_out, int out_size, void* d_ws, size_t ws_size,
                              hipStream_t stream) {
    const float* x  = (const float*)d_in[0];
    const float* h0 = (const float*)d_in[1];
    const float* Wx = (const float*)d_in[2];
    const float* Wh = (const float*)d_in[3];
    const float* bv = (const float*)d_in[4];

    char* ws = (char*)d_ws;
    bf16_t* WxT  = (bf16_t*)(ws);                                  // 1 MB
    bf16_t* WhT  = (bf16_t*)(ws + (size_t)(1u << 20));             // 2 MB
    bf16_t* hbf0 = (bf16_t*)(ws + (size_t)(3u << 20));             // 128 KB
    bf16_t* hbf1 = (bf16_t*)(ws + (size_t)(3u << 20) + (1u << 18));// 128 KB
    float*  xw   = (float*)(ws + (size_t)(4u << 20));              // 128 MB

    // weight conversion / transposition (one-time)
    cvt_transpose_kernel<<<(kI * kH + 255) / 256, 256, 0, stream>>>(Wx, WxT, kI, kH);
    cvt_transpose_kernel<<<(kH * kH + 255) / 256, 256, 0, stream>>>(Wh, WhT, kH, kH);
    cvt_kernel<<<(kB * kH + 255) / 256, 256, 0, stream>>>(h0, hbf0, kB * kH);

    // Phase 1: xw = x @ Wx + b  (131072 tiles / 8 waves per block)
    {
        int tiles = (kT * kB / 16) * (kH / 16);
        xw_gemm_kernel<<<tiles / 8, 256, 0, stream>>>(x, WxT, bv, xw);
    }

    // Phase 2: sequential scan, one kernel per timestep (stream-ordered).
    float* out = (float*)d_out;
    bf16_t* hp[2] = {hbf0, hbf1};
    for (int t = 0; t < kT; ++t) {
        rnn_step_kernel<<<(4 * 64) / 8, 256, 0, stream>>>(
            xw + (size_t)t * kB * kH, WhT,
            hp[t & 1], hp[(t + 1) & 1],
            out + (size_t)t * kB * kH);
    }

    // h_last = outs[T-1]
    hipMemcpyAsync(out + (size_t)kT * kB * kH,
                   out + (size_t)(kT - 1) * kB * kH,
                   (size_t)kB * kH * sizeof(float),
                   hipMemcpyDeviceToDevice, stream);
}
